// MultiHeadLatentAttention_29643864277019
// MI455X (gfx1250) — matrix operand order repaired and checked by reading, not compiled, not run on hardware
//
#include <hip/hip_runtime.h>

typedef __attribute__((ext_vector_type(16))) __bf16 bf16x16;
typedef __attribute__((ext_vector_type(8)))  __bf16 bf16x8;
typedef __attribute__((ext_vector_type(8)))  float  f32x8;

namespace {
constexpr int kD   = 1024;   // model dim
constexpr int kH   = 16;     // heads
constexpr int kDh  = 64;     // head dim
constexpr int kDr  = 32;     // rot dim
constexpr int kDqk = 96;     // Dh + Dr
constexpr int kB   = 2;
constexpr int kS   = 2048;
constexpr int kM   = kB * kS;                 // 4096 token rows
constexpr int kNP  = 3 * kD + kH * kDr;       // 3584 fused projection cols
constexpr float kScale = 0.10206207261596577f; // 1/sqrt(96)
}

__device__ __forceinline__ f32x8 wmma_bf16(bf16x16 a, bf16x16 b, f32x8 c) {
  return __builtin_amdgcn_wmma_f32_16x16x32_bf16(false, a, false, b, (short)0, c,
                                                 false, false);
}

// A-fragment: 16x32 bf16 tile at row-major (row0,k0), leading dim lda.
// Lanes 0-15: M=lane, K 0-15; lanes 16-31: M=lane-16, K 16-31.
__device__ __forceinline__ bf16x16 load_a(const __bf16* a00, int lda, int lane) {
  const int m = lane & 15, half = lane >> 4;
  const __bf16* p = a00 + (size_t)m * lda + half * 16;
  bf16x8 lo = *(const bf16x8*)(p);
  bf16x8 hi = *(const bf16x8*)(p + 8);
  bf16x16 r;
#pragma unroll
  for (int i = 0; i < 8; ++i) { r[i] = lo[i]; r[i + 8] = hi[i]; }
  return r;
}

// B-fragment: 32x16 tile; Bt stored [N x K] row-major (B column-major).
// Lane = column n, element e -> K = (lane>>4)*16 + e : one contiguous 32B load.
__device__ __forceinline__ bf16x16 load_b(const __bf16* bt00, int ldb, int lane) {
  const int n = lane & 15, kh = (lane >> 4) * 16;
  return *(const bf16x16*)(bt00 + (size_t)n * ldb + kh);
}

// ---------------------------------------------------------------------------
__global__ void k_cvt_bf16(const float* __restrict__ src, __bf16* __restrict__ dst,
                           size_t n) {
  for (size_t i = blockIdx.x * (size_t)blockDim.x + threadIdx.x; i < n;
       i += (size_t)gridDim.x * blockDim.x)
    dst[i] = (__bf16)src[i];
}

// transpose + convert: src f32 [K x N] -> dst bf16 [N x K]
__global__ void k_transpose_bf16(const float* __restrict__ src,
                                 __bf16* __restrict__ dst, int K, int N) {
  const size_t total = (size_t)K * N;
  for (size_t i = blockIdx.x * (size_t)blockDim.x + threadIdx.x; i < total;
       i += (size_t)gridDim.x * blockDim.x) {
    const int n = (int)(i % N);
    const int k = (int)(i / N);
    dst[(size_t)n * K + k] = (__bf16)src[i];
  }
}

__global__ void k_pack_bias(const float* __restrict__ bq, const float* __restrict__ bk,
                            const float* __restrict__ bv, const float* __restrict__ br,
                            float* __restrict__ ball) {
  const int i = blockIdx.x * blockDim.x + threadIdx.x;
  if (i < kD)               ball[i] = bq[i];
  else if (i < 2 * kD)      ball[i] = bk[i - kD];
  else if (i < 3 * kD)      ball[i] = bv[i - 2 * kD];
  else if (i < kNP)         ball[i] = br[i - 3 * kD];
}

// ---------------------------------------------------------------------------
// bf16 GEMM: C[M x N] = A[M x K] * Bt[N x K]^T + bias.
// 256 threads = 8 waves (4 in M x 2 in N); block tile 128x128; wave tile 32x64.
__global__ void __launch_bounds__(256)
k_gemm_bf16(const __bf16* __restrict__ A, const __bf16* __restrict__ Bt,
            const float* __restrict__ bias, __bf16* __restrict__ Cb,
            float* __restrict__ Cf, int M, int N, int K, int storeF32) {
  (void)M;
  const int lane = threadIdx.x & 31;
  const int wave = threadIdx.x >> 5;
  const int wm = wave & 3, wn = wave >> 2;
  const int ntile = N >> 7;
  const int m0 = (blockIdx.x / ntile) * 128 + wm * 32;
  const int n0 = (blockIdx.x % ntile) * 128 + wn * 64;
  const int col = lane & 15, half = lane >> 4;

  f32x8 acc[2][4];
#pragma unroll
  for (int ms = 0; ms < 2; ++ms)
#pragma unroll
    for (int t = 0; t < 4; ++t)
#pragma unroll
      for (int r = 0; r < 8; ++r) acc[ms][t][r] = 0.f;

  const __bf16* a00 = A + (size_t)m0 * K;
  for (int k = 0; k < K; k += 32) {
    if (k + 128 < K) {
      __builtin_prefetch(A + (size_t)(m0 + col) * K + k + 128, 0, 0);
      __builtin_prefetch(Bt + (size_t)(n0 + col) * K + k + 128, 0, 0);
    }
    const bf16x16 a0 = load_a(a00 + k, K, lane);
    const bf16x16 a1 = load_a(a00 + (size_t)16 * K + k, K, lane);
#pragma unroll
    for (int t = 0; t < 4; ++t) {
      const bf16x16 b = load_b(Bt + (size_t)(n0 + t * 16) * K + k, K, lane);
      acc[0][t] = wmma_bf16(a0, b, acc[0][t]);
      acc[1][t] = wmma_bf16(a1, b, acc[1][t]);
    }
  }
#pragma unroll
  for (int ms = 0; ms < 2; ++ms) {
#pragma unroll
    for (int t = 0; t < 4; ++t) {
      const int n = n0 + t * 16 + col;
      const float bv = bias ? bias[n] : 0.f;
#pragma unroll
      for (int r = 0; r < 8; ++r) {
        const int m = m0 + ms * 16 + r + 8 * half;  // C layout: M = vgpr + 8*half
        const float v = acc[ms][t][r] + bv;
        if (storeF32) Cf[(size_t)m * N + n] = v;
        else          Cb[(size_t)m * N + n] = (__bf16)v;
      }
    }
  }
}

// ---------------------------------------------------------------------------
// Rearrange fused projections P [kM x 3584] into attention layouts:
//   Qb [B,H,S,96] = concat(q, r) * scale
//   Kb [B,H,S,96] = concat(k, r)
//   Vt [B,H,64,S] = v transposed per head
__global__ void k_scatter(const __bf16* __restrict__ P, __bf16* __restrict__ Qb,
                          __bf16* __restrict__ Kb, __bf16* __restrict__ Vt) {
  const size_t stride = (size_t)gridDim.x * blockDim.x;
  const size_t nQK = (size_t)kM * kH * kDqk;
  for (size_t i = blockIdx.x * (size_t)blockDim.x + threadIdx.x; i < nQK; i += stride) {
    const int d = (int)(i % kDqk);
    const size_t t = i / kDqk;
    const int h = (int)(t % kH);
    const size_t bs = t / kH;                       // 0..4095
    const int b = (int)(bs >> 11), s = (int)(bs & 2047);
    float qv, kv;
    if (d < kDh) {
      qv = (float)P[bs * kNP + h * kDh + d];
      kv = (float)P[bs * kNP + kD + h * kDh + d];
    } else {
      const float rv = (float)P[bs * kNP + 3 * kD + h * kDr + (d - kDh)];
      qv = rv; kv = rv;
    }
    const size_t dst = (((size_t)(b * kH + h)) * kS + s) * kDqk + d;
    Qb[dst] = (__bf16)(qv * kScale);
    Kb[dst] = (__bf16)kv;
  }
  const size_t nV = (size_t)kM * kD;
  for (size_t i = blockIdx.x * (size_t)blockDim.x + threadIdx.x; i < nV; i += stride) {
    const int c = (int)(i % kD);
    const size_t bs = i / kD;
    const int h = c >> 6, d = c & 63;
    const int b = (int)(bs >> 11), s = (int)(bs & 2047);
    Vt[(((size_t)(b * kH + h)) * kDh + d) * kS + s] = P[bs * kNP + 2 * kD + c];
  }
}

// ---------------------------------------------------------------------------
// Flash attention: 4 waves/block, 32 queries/wave (128 q/block), 32-key blocks.
// K tile (32x96) and V tile (64x32, transposed V) staged in LDS and shared by
// all 4 waves; per-wave P tile bounced through LDS for the C->A relayout.
__global__ void __launch_bounds__(128)
k_flash_attn(const __bf16* __restrict__ Qb, const __bf16* __restrict__ Kb,
             const __bf16* __restrict__ Vt, __bf16* __restrict__ Ob) {
  __shared__ __bf16 ldsK[32 * kDqk];    // 6 KB  : keys kb..kb+31, 96 dims
  __shared__ __bf16 ldsV[kDh * 32];     // 4 KB  : [d][key] tile
  __shared__ __bf16 ldsP[4 * 16 * 40];  // 5 KB  : per-wave P bounce, stride 40
  const int tid = threadIdx.x;
  const int lane = tid & 31;
  const int wave = tid >> 5;
  const int col = lane & 15, half = lane >> 4;
  const int qblocks = kS / 128;
  const int bh = blockIdx.x / qblocks;  // b*16 + h
  const int qt = blockIdx.x % qblocks;
  const int q0 = qt * 128 + wave * 32;  // this wave's 32 queries
  const __bf16* Qh = Qb + (size_t)bh * kS * kDqk;
  const __bf16* Kh = Kb + (size_t)bh * kS * kDqk;
  const __bf16* Vh = Vt + (size_t)bh * kDh * kS;
  __bf16* lw = ldsP + wave * (16 * 40);

  bf16x16 qa[2][3];                     // 2 q-subtiles x 96 dims (3 k-steps)
#pragma unroll
  for (int qs = 0; qs < 2; ++qs)
#pragma unroll
    for (int s3 = 0; s3 < 3; ++s3)
      qa[qs][s3] = load_a(Qh + (size_t)(q0 + qs * 16) * kDqk + s3 * 32, kDqk, lane);

  float m_r[2][8], l_r[2][8];
  f32x8 acc[2][4];
#pragma unroll
  for (int qs = 0; qs < 2; ++qs)
#pragma unroll
    for (int r = 0; r < 8; ++r) { m_r[qs][r] = -1e30f; l_r[qs][r] = 0.f; }
#pragma unroll
  for (int qs = 0; qs < 2; ++qs)
#pragma unroll
    for (int t = 0; t < 4; ++t)
#pragma unroll
      for (int r = 0; r < 8; ++r) acc[qs][t][r] = 0.f;

  for (int kb = 0; kb < kS; kb += 32) {
    // ---- cooperative staging: K rows kb..kb+31 are contiguous (3072 elems)
    {
      const __bf16* src = Kh + (size_t)kb * kDqk;
#pragma unroll
      for (int j = 0; j < 3; ++j) {
        const int idx = j * 1024 + tid * 8;
        *(bf16x8*)(ldsK + idx) = *(const bf16x8*)(src + idx);
      }
#pragma unroll
      for (int j = 0; j < 2; ++j) {
        const int c = j * 128 + tid;          // 256 chunks: 64 rows x 4 parts
        const int row = c >> 2, part = c & 3;
        *(bf16x8*)(ldsV + row * 32 + part * 8) =
            *(const bf16x8*)(Vh + (size_t)row * kS + kb + part * 8);
      }
    }
    __syncthreads();
#pragma unroll
    for (int qs = 0; qs < 2; ++qs) {
      // ---- scores: two 16-key tiles, 3 WMMAs each over the 96-dim
      f32x8 s0, s1;
#pragma unroll
      for (int sub = 0; sub < 2; ++sub) {
        const __bf16* kp = ldsK + (sub * 16 + col) * kDqk + half * 16;
        f32x8 st;
#pragma unroll
        for (int r = 0; r < 8; ++r) st[r] = 0.f;
        st = wmma_bf16(qa[qs][0], *(const bf16x16*)(kp),      st);
        st = wmma_bf16(qa[qs][1], *(const bf16x16*)(kp + 32), st);
        st = wmma_bf16(qa[qs][2], *(const bf16x16*)(kp + 64), st);
        if (sub == 0) s0 = st; else s1 = st;
      }
      // ---- online softmax; accumulator row r = query (r + 8*half)
#pragma unroll
      for (int r = 0; r < 8; ++r) {
        float mx = fmaxf(s0[r], s1[r]);
#pragma unroll
        for (int msk = 1; msk < 16; msk <<= 1) mx = fmaxf(mx, __shfl_xor(mx, msk, 32));
        const float mn = fmaxf(m_r[qs][r], mx);
        const float p0 = __expf(s0[r] - mn);
        const float p1 = __expf(s1[r] - mn);
        const float corr = __expf(m_r[qs][r] - mn);
        float rs = p0 + p1;
#pragma unroll
        for (int msk = 1; msk < 16; msk <<= 1) rs += __shfl_xor(rs, msk, 32);
        l_r[qs][r] = l_r[qs][r] * corr + rs;
        m_r[qs][r] = mn;
#pragma unroll
        for (int t = 0; t < 4; ++t) acc[qs][t][r] *= corr;
        const int row = r + 8 * half;
        lw[row * 40 + col]      = (__bf16)p0;   // keys kb + 0..15
        lw[row * 40 + 16 + col] = (__bf16)p1;   // keys kb + 16..31
      }
      asm volatile("s_wait_dscnt 0" ::: "memory");
      // ---- reload P as A-fragment (16 q x 32 keys)
      bf16x16 pa;
      {
        const __bf16* p = lw + (size_t)col * 40 + half * 16;
        bf16x8 lo = *(const bf16x8*)(p);
        bf16x8 hi = *(const bf16x8*)(p + 8);
#pragma unroll
        for (int i = 0; i < 8; ++i) { pa[i] = lo[i]; pa[i + 8] = hi[i]; }
      }
      // ---- O += P(16x32) * V(32x64): 4 WMMAs from the shared V tile
#pragma unroll
      for (int t = 0; t < 4; ++t) {
        const __bf16* vp = ldsV + (t * 16 + col) * 32 + half * 16;
        acc[qs][t] = wmma_bf16(pa, *(const bf16x16*)(vp), acc[qs][t]);
      }
      asm volatile("s_wait_dscnt 0" ::: "memory");  // lw reused by next qs
    }
    __syncthreads();  // ldsK/ldsV reused next key block
  }

  // ---- normalize + store token-major bf16 [kM x kD]
  const int b = bh >> 4, h = bh & 15;
#pragma unroll
  for (int qs = 0; qs < 2; ++qs)
#pragma unroll
    for (int t = 0; t < 4; ++t)
#pragma unroll
      for (int r = 0; r < 8; ++r) {
        const int q = q0 + qs * 16 + r + 8 * half;
        const float v = acc[qs][t][r] / l_r[qs][r];
        Ob[((size_t)(b * kS + q)) * kD + h * kDh + t * 16 + col] = (__bf16)v;
      }
}

// ---------------------------------------------------------------------------
extern "C" void kernel_launch(void* const* d_in, const int* in_sizes, int n_in,
                              void* d_out, int out_size, void* d_ws, size_t ws_size,
                              hipStream_t stream) {
  (void)in_sizes; (void)n_in; (void)out_size; (void)ws_size;
  const float* x  = (const float*)d_in[0];
  const float* Wq = (const float*)d_in[1];
  const float* bq = (const float*)d_in[2];
  const float* Wk = (const float*)d_in[3];
  const float* bk = (const float*)d_in[4];
  const float* Wv = (const float*)d_in[5];
  const float* bv = (const float*)d_in[6];
  const float* Wr = (const float*)d_in[7];
  const float* br = (const float*)d_in[8];
  const float* Wo = (const float*)d_in[9];
  const float* bo = (const float*)d_in[10];
  float* out = (float*)d_out;

  char* ws = (char*)d_ws;
  size_t off = 0;
  auto take = [&](size_t bytes) {
    char* p = ws + off;
    off += (bytes + 255) & ~(size_t)255;
    return p;
  };
  __bf16* xb   = (__bf16*)take((size_t)kM * kD * 2);
  __bf16* Wall = (__bf16*)take((size_t)kNP * kD * 2);   // [3584 x 1024] = Wq|Wk|Wv|Wr ^T
  __bf16* Wot  = (__bf16*)take((size_t)kD * kD * 2);
  float*  ball = (float*)take((size_t)kNP * 4);
  __bf16* P    = (__bf16*)take((size_t)kM * kNP * 2);   // fused projections
  __bf16* Qb   = (__bf16*)take((size_t)kB * kH * kS * kDqk * 2);
  __bf16* Kb   = (__bf16*)take((size_t)kB * kH * kS * kDqk * 2);
  __bf16* Vtb  = (__bf16*)take((size_t)kB * kH * kDh * kS * 2);
  __bf16* Ob   = (__bf16*)take((size_t)kM * kD * 2);

  // 1) precision conversion + packed weight transposes
  k_cvt_bf16<<<1024, 256, 0, stream>>>(x, xb, (size_t)kM * kD);
  k_transpose_bf16<<<512, 256, 0, stream>>>(Wq, Wall,                        kD, kD);
  k_transpose_bf16<<<512, 256, 0, stream>>>(Wk, Wall + (size_t)kD * kD,      kD, kD);
  k_transpose_bf16<<<512, 256, 0, stream>>>(Wv, Wall + (size_t)2 * kD * kD,  kD, kD);
  k_transpose_bf16<<<512, 256, 0, stream>>>(Wr, Wall + (size_t)3 * kD * kD,  kD, kH * kDr);
  k_transpose_bf16<<<512, 256, 0, stream>>>(Wo, Wot, kD, kD);
  k_pack_bias<<<(kNP + 255) / 256, 256, 0, stream>>>(bq, bk, bv, br, ball);

  // 2) fused projection GEMM: [4096 x 1024] x [1024 x 3584] -> bf16
  k_gemm_bf16<<<(kM / 128) * (kNP / 128), 256, 0, stream>>>(
      xb, Wall, ball, P, nullptr, kM, kNP, kD, 0);

  // 3) head-major rearrangement (Q pre-scaled, V transposed)
  k_scatter<<<2048, 256, 0, stream>>>(P, Qb, Kb, Vtb);

  // 4) flash attention (LDS-shared K/V tiles, 128 queries per block)
  k_flash_attn<<<kB * kH * (kS / 128), 128, 0, stream>>>(Qb, Kb, Vtb, Ob);

  // 5) output projection, f32 result to d_out
  k_gemm_bf16<<<(kM / 128) * (kD / 128), 256, 0, stream>>>(
      Ob, Wot, bo, nullptr, out, kM, kD, kD, 1);
}
